// dynamic_graph_69123203661950
// MI455X (gfx1250) — matrix operand order repaired and checked by reading, not compiled, not run on hardware
//
#include <hip/hip_runtime.h>
#include <math.h>

// ---------------------------------------------------------------------------
// CDNA5 (gfx1250) dynamic-graph attention pipeline, v3.
// - all GEMMs on v_wmma_f32_16x16x32_bf16 (wave32)
// - branchless fragment loaders (padded buffers), b128 loads
// - B operands pre-packed to bf16 fragment layout (single 32B load/lane)
// - register double-buffering in all streaming loops to hide load latency
// - flash-style recompute: attention matrices never materialized in f32
// ---------------------------------------------------------------------------

typedef __bf16 bf16_t;
typedef __attribute__((ext_vector_type(16))) __bf16 v16bf;
typedef __attribute__((ext_vector_type(8)))  __bf16 v8bf;
typedef __attribute__((ext_vector_type(8)))  float  v8f;

#define NQ    8000
#define NKN   8000
#define DIMF  256
#define DH    64          // dim/4
#define D2    32          // dim/8
#define EDGE  3201        // int(8000*0.4)+1
#define NN    (NQ + EDGE) // 11201
#define NNP   11232       // NN padded to multiple of 32
#define SCALE 0.0625f     // 1/sqrt(256)
#define LN_EPS 1e-5f
#define STDF  0.2f
#define KW    0.3f

static __device__ __forceinline__ bf16_t f2bf(float x) { return (bf16_t)x; }

static __device__ __forceinline__ v8f wmma_bf(v16bf a, v16bf b, v8f c) {
  return __builtin_amdgcn_wmma_f32_16x16x32_bf16(false, a, false, b, (short)0, c,
                                                 false, false);
}

// ---- branchless fragment loaders (ISA 7.12.2 layouts) ----------------------
static __device__ __forceinline__ v16bf
load_a_rm(const float* __restrict__ X, int ld, int row0, int k0) {
  const int lane = threadIdx.x & 31;
  const int m = lane & 15, g = lane >> 4;
  const float4* p = (const float4*)(X + (size_t)(row0 + m) * ld + k0 + 8 * g);
  float4 c0 = p[0], c1 = p[1], c2 = p[4], c3 = p[5];
  v16bf a;
  a[0] = f2bf(c0.x); a[1] = f2bf(c0.y); a[2]  = f2bf(c0.z); a[3]  = f2bf(c0.w);
  a[4] = f2bf(c1.x); a[5] = f2bf(c1.y); a[6]  = f2bf(c1.z); a[7]  = f2bf(c1.w);
  a[8] = f2bf(c2.x); a[9] = f2bf(c2.y); a[10] = f2bf(c2.z); a[11] = f2bf(c2.w);
  a[12]= f2bf(c3.x); a[13]= f2bf(c3.y); a[14] = f2bf(c3.z); a[15] = f2bf(c3.w);
  return a;
}

static __device__ __forceinline__ v16bf
load_a_lds32(const float* P, int ld, int k0) {
  const int lane = threadIdx.x & 31;
  const int m = lane & 15, g = lane >> 4;
  const float4* p = (const float4*)(P + m * ld + k0 + 8 * g);
  float4 c0 = p[0], c1 = p[1], c2 = p[4], c3 = p[5];
  v16bf a;
  a[0] = f2bf(c0.x); a[1] = f2bf(c0.y); a[2]  = f2bf(c0.z); a[3]  = f2bf(c0.w);
  a[4] = f2bf(c1.x); a[5] = f2bf(c1.y); a[6]  = f2bf(c1.z); a[7]  = f2bf(c1.w);
  a[8] = f2bf(c2.x); a[9] = f2bf(c2.y); a[10] = f2bf(c2.z); a[11] = f2bf(c2.w);
  a[12]= f2bf(c3.x); a[13]= f2bf(c3.y); a[14] = f2bf(c3.z); a[15] = f2bf(c3.w);
  return a;
}

// A fragment from LDS-resident bf16 16x32 P-tile (attn2 flash step)
static __device__ __forceinline__ v16bf load_a_pt(const bf16_t* P) {
  const int lane = threadIdx.x & 31;
  const int m = lane & 15, g = lane >> 4;
  const bf16_t* pp = P + m * 32 + 8 * g;
  v8bf lo = *(const v8bf*)(pp);
  v8bf hi = *(const v8bf*)(pp + 16);
  return __builtin_shufflevector(lo, hi, 0, 1, 2, 3, 4, 5, 6, 7,
                                 8, 9, 10, 11, 12, 13, 14, 15);
}

// pre-packed B fragment: one 32B contiguous load per lane
static __device__ __forceinline__ v16bf
load_frag(const bf16_t* __restrict__ F, int tile) {
  const int lane = threadIdx.x & 31;
  return *(const v16bf*)(F + ((size_t)tile * 32 + lane) * 16);
}

// ---- one-shot packers into B fragment layout -------------------------------
__global__ __launch_bounds__(32) void pack_b_kn(
    const float* __restrict__ Wm, int K, int N, bf16_t* __restrict__ out) {
  const int lane = threadIdx.x & 31;
  const int KT = K >> 5;
  const int tile = blockIdx.x;
  const int kt = tile % KT, nt = tile / KT;
  const int n = nt * 16 + (lane & 15), g = lane >> 4;
  const float* src = Wm + (size_t)(kt * 32 + 8 * g) * N + n;
  v16bf b;
#pragma unroll
  for (int h = 0; h < 16; ++h) b[h] = f2bf(src[(size_t)(h < 8 ? h : h + 8) * N]);
  *(v16bf*)(out + ((size_t)tile * 32 + lane) * 16) = b;
}

__global__ __launch_bounds__(32) void pack_b_nk(
    const float* __restrict__ S, int ld, bf16_t* __restrict__ out) {
  const int lane = threadIdx.x & 31;
  const int KT = ld >> 5;
  const int tile = blockIdx.x;
  const int kt = tile % KT, jt = tile / KT;
  const int n = jt * 16 + (lane & 15), g = lane >> 4;
  const float4* src = (const float4*)(S + (size_t)n * ld + kt * 32 + 8 * g);
  float4 c0 = src[0], c1 = src[1], c2 = src[4], c3 = src[5];
  v16bf b;
  b[0] = f2bf(c0.x); b[1] = f2bf(c0.y); b[2]  = f2bf(c0.z); b[3]  = f2bf(c0.w);
  b[4] = f2bf(c1.x); b[5] = f2bf(c1.y); b[6]  = f2bf(c1.z); b[7]  = f2bf(c1.w);
  b[8] = f2bf(c2.x); b[9] = f2bf(c2.y); b[10] = f2bf(c2.z); b[11] = f2bf(c2.w);
  b[12]= f2bf(c3.x); b[13]= f2bf(c3.y); b[14] = f2bf(c3.z); b[15] = f2bf(c3.w);
  *(v16bf*)(out + ((size_t)tile * 32 + lane) * 16) = b;
}

// ---- fused MLP2: Y = relu(X@W1+b1)@W2+b2, compile-time shapes --------------
template <int K1, int H, int K2>
__global__ __launch_bounds__(32) void mlp2_wmma(
    const float* __restrict__ X, int nrows,
    const bf16_t* __restrict__ W1f, const float* __restrict__ B1,
    const bf16_t* __restrict__ W2f, const float* __restrict__ B2,
    float* __restrict__ Y) {
  constexpr int KT1 = K1 / 32, NT1 = H / 16, KT2 = H / 32, NT2 = K2 / 16;
  __shared__ float hid[16 * H];
  const int lane = threadIdx.x & 31;
  const int n = lane & 15, g = lane >> 4;
  const int row0 = blockIdx.x * 16;

  v16bf af[KT1];
#pragma unroll
  for (int kt = 0; kt < KT1; ++kt) af[kt] = load_a_rm(X, K1, row0, kt * 32);

#pragma unroll
  for (int nt = 0; nt < NT1; ++nt) {
    v8f acc = {};
#pragma unroll
    for (int kt = 0; kt < KT1; ++kt)
      acc = wmma_bf(af[kt], load_frag(W1f, nt * KT1 + kt), acc);
    float bias = B1[nt * 16 + n];
#pragma unroll
    for (int r = 0; r < 8; ++r) {
      float v = acc[r] + bias;
      hid[(r + 8 * g) * H + nt * 16 + n] = v > 0.0f ? v : 0.0f;
    }
  }
  __syncthreads();
#pragma unroll
  for (int nt = 0; nt < NT2; ++nt) {
    v8f acc = {};
#pragma unroll
    for (int kt = 0; kt < KT2; ++kt)
      acc = wmma_bf(load_a_lds32(hid, H, kt * 32), load_frag(W2f, nt * KT2 + kt), acc);
    float bias = B2[nt * 16 + n];
#pragma unroll
    for (int r = 0; r < 8; ++r) {
      int row = row0 + r + 8 * g;
      if (row < nrows) Y[(size_t)row * K2 + nt * 16 + n] = acc[r] + bias;
    }
  }
}

// ---- attn1 pass A: per-row online softmax stats (double-buffered B) --------
__global__ __launch_bounds__(32) void attn1_rowstats(
    const float* __restrict__ qh, const bf16_t* __restrict__ khf,
    float* __restrict__ rowmax, float* __restrict__ rowsum) {
  __shared__ float tile[16][16];
  const int lane = threadIdx.x & 31;
  const int n = lane & 15, g = lane >> 4;
  const int i0 = blockIdx.x * 16;
  constexpr int JT = NKN / 16;
  v16bf a0 = load_a_rm(qh, DH, i0, 0);
  v16bf a1 = load_a_rm(qh, DH, i0, 32);
  v16bf b0 = load_frag(khf, 0), b1 = load_frag(khf, 1);
  float m = -__builtin_inff(), s = 0.0f;            // lane<16 owns row i0+lane
  for (int jt = 0; jt < JT; ++jt) {
    const int jn = (jt + 1 < JT) ? jt + 1 : jt;     // prefetch next (clamped)
    v16bf nb0 = load_frag(khf, jn * 2 + 0);
    v16bf nb1 = load_frag(khf, jn * 2 + 1);
    v8f acc = {};
    acc = wmma_bf(a0, b0, acc);
    acc = wmma_bf(a1, b1, acc);
#pragma unroll
    for (int r = 0; r < 8; ++r) tile[r + 8 * g][n] = acc[r] * SCALE;
    __syncthreads();
    if (lane < 16) {
#pragma unroll
      for (int c = 0; c < 16; ++c) {
        float l = tile[lane][c];
        float mn = fmaxf(m, l);
        s = s * __expf(m - mn) + __expf(l - mn);
        m = mn;
      }
    }
    __syncthreads();
    b0 = nb0; b1 = nb1;
  }
  if (lane < 16) { rowmax[i0 + lane] = m; rowsum[i0 + lane] = s; }
}

// ---- attn1 pass B: score[j] = max_i softmax_row(i)[j] ----------------------
__global__ __launch_bounds__(32) void attn1_score(
    const float* __restrict__ qh, const bf16_t* __restrict__ khf,
    const float* __restrict__ rowmax, const float* __restrict__ rowsum,
    unsigned* __restrict__ scoreBits) {
  __shared__ float tile[16][16];
  __shared__ float srm[16], srs[16];
  const int lane = threadIdx.x & 31;
  const int n = lane & 15, g = lane >> 4;
  const int i0 = blockIdx.x * 16;
  constexpr int JT = NKN / 16;
  if (lane < 16) { srm[lane] = rowmax[i0 + lane]; srs[lane] = rowsum[i0 + lane]; }
  v16bf a0 = load_a_rm(qh, DH, i0, 0);
  v16bf a1 = load_a_rm(qh, DH, i0, 32);
  v16bf b0 = load_frag(khf, 0), b1 = load_frag(khf, 1);
  __syncthreads();
  for (int jt = 0; jt < JT; ++jt) {
    const int jn = (jt + 1 < JT) ? jt + 1 : jt;
    v16bf nb0 = load_frag(khf, jn * 2 + 0);
    v16bf nb1 = load_frag(khf, jn * 2 + 1);
    v8f acc = {};
    acc = wmma_bf(a0, b0, acc);
    acc = wmma_bf(a1, b1, acc);
#pragma unroll
    for (int r = 0; r < 8; ++r) tile[r + 8 * g][n] = acc[r] * SCALE;
    __syncthreads();
    if (lane < 16) {                                 // lane owns column jt*16+lane
      float cm = 0.0f;
#pragma unroll
      for (int r = 0; r < 16; ++r)
        cm = fmaxf(cm, __expf(tile[r][lane] - srm[r]) / srs[r]);
      atomicMax(&scoreBits[jt * 16 + lane], __float_as_uint(cm));
    }
    __syncthreads();
    b0 = nb0; b1 = nb1;
  }
}

// ---- node assembly ---------------------------------------------------------
__global__ void copy_qh(const float* __restrict__ qh, float* __restrict__ node) {
  int i = blockIdx.x * blockDim.x + threadIdx.x;
  if (i < NQ * DH) node[i] = qh[i];
}

__global__ __launch_bounds__(256) void select_topk(
    const float* __restrict__ score, const float* __restrict__ kh,
    float* __restrict__ node) {
  __shared__ float s[NKN];                           // 32KB of 320KB WGP LDS
  for (int t = threadIdx.x; t < NKN; t += 256) s[t] = score[t];
  __syncthreads();
  int j = blockIdx.x * 256 + threadIdx.x;
  if (j >= NKN) return;
  float sj = s[j];
  int rank = 0;
  for (int t = 0; t < NKN; ++t) {
    float st = s[t];
    rank += (st > sj) || (st == sj && t < j);
  }
  if (rank < EDGE) {
    float* dst = node + (size_t)(NQ + rank) * DH;
    const float* src = kh + (size_t)j * DH;
    for (int c = 0; c < DH; ++c) dst[c] = KW * src[c];
  }
}

// ---- attn2 column stats (softmax over dim 0), double-buffered A ------------
__global__ __launch_bounds__(32) void attn2_colstats(
    const float* __restrict__ q2, const bf16_t* __restrict__ k2f,
    float* __restrict__ colmax, float* __restrict__ colsum) {
  __shared__ float tile[16][16];
  const int lane = threadIdx.x & 31;
  const int n = lane & 15, g = lane >> 4;
  const int jt = blockIdx.x;
  v16bf b = load_frag(k2f, jt);
  v16bf a = load_a_rm(q2, D2, 0, 0);
  float m = -__builtin_inff(), s = 0.0f;             // lane<16 owns col jt*16+lane
  for (int i0 = 0; i0 < NN; i0 += 16) {
    const int in = (i0 + 16 < NN) ? i0 + 16 : i0;
    v16bf na = load_a_rm(q2, D2, in, 0);
    v8f acc = {};
    acc = wmma_bf(a, b, acc);
#pragma unroll
    for (int r = 0; r < 8; ++r) tile[r + 8 * g][n] = acc[r] * SCALE;
    __syncthreads();
    if (lane < 16) {
      int rv = NN - i0; if (rv > 16) rv = 16;
      for (int r = 0; r < rv; ++r) {
        float l = tile[r][lane];
        float mn = fmaxf(m, l);
        s = s * __expf(m - mn) + __expf(l - mn);
        m = mn;
      }
    }
    __syncthreads();
    a = na;
  }
  if (lane < 16) {
    int j = jt * 16 + lane;
    bool ok = j < NN;                                // padding -> identity stats
    colmax[j] = ok ? m : 0.0f;
    colsum[j] = ok ? s : 1.0f;
  }
}

// ---- fused attn2 @ node (+residual) and sum(p^2), double-buffered ----------
__global__ __launch_bounds__(32) void attn2_apply(
    const float* __restrict__ q2, const bf16_t* __restrict__ k2f,
    const bf16_t* __restrict__ nodef, const float* __restrict__ nodeIn,
    const float* __restrict__ colmax, const float* __restrict__ colsum,
    float* __restrict__ node2, float* __restrict__ sumsq) {
  __shared__ bf16_t Pt[16 * 32];
  __shared__ float red[32];
  const int lane = threadIdx.x & 31;
  const int n = lane & 15, g = lane >> 4;
  const int i0 = blockIdx.x * 16;
  constexpr int KTN = NNP / 32;                      // node fragment K-tiles
  v16bf aq = load_a_rm(q2, D2, i0, 0);
  v8f o0 = {}, o1 = {}, o2 = {}, o3 = {};
  float ssq = 0.0f;
  // current tile fragments
  v16bf bk0 = load_frag(k2f, 0), bk1 = load_frag(k2f, 1);
  v16bf bn0 = load_frag(nodef, 0 * KTN);
  v16bf bn1 = load_frag(nodef, 1 * KTN);
  v16bf bn2 = load_frag(nodef, 2 * KTN);
  v16bf bn3 = load_frag(nodef, 3 * KTN);
  for (int j0 = 0; j0 < NNP; j0 += 32) {
    const int kt = j0 >> 5;
    const int kn = (kt + 1 < KTN) ? kt + 1 : kt;     // prefetch next (clamped)
    v16bf nk0 = load_frag(k2f, 2 * kn + 0);
    v16bf nk1 = load_frag(k2f, 2 * kn + 1);
    v16bf nn0 = load_frag(nodef, 0 * KTN + kn);
    v16bf nn1 = load_frag(nodef, 1 * KTN + kn);
    v16bf nn2 = load_frag(nodef, 2 * KTN + kn);
    v16bf nn3 = load_frag(nodef, 3 * KTN + kn);
    v8f d0 = {}, d1 = {};
    d0 = wmma_bf(aq, bk0, d0);
    d1 = wmma_bf(aq, bk1, d1);
#pragma unroll
    for (int js = 0; js < 2; ++js) {
      const v8f& d = js ? d1 : d0;
      const int j = j0 + 16 * js + n;
      const float cm = colmax[j], cs = colsum[j];    // padded -> (0,1)
      const bool jok = j < NN;
#pragma unroll
      for (int r = 0; r < 8; ++r) {
        float p = __expf(d[r] * SCALE - cm) / cs;
        bool ok = jok && (i0 + r + 8 * g) < NN;
        ssq += ok ? p * p : 0.0f;                    // branchless select
        Pt[(r + 8 * g) * 32 + 16 * js + n] = f2bf(p); // padded j: p*node==0
      }
    }
    __syncthreads();
    v16bf ap = load_a_pt(Pt);
    o0 = wmma_bf(ap, bn0, o0);
    o1 = wmma_bf(ap, bn1, o1);
    o2 = wmma_bf(ap, bn2, o2);
    o3 = wmma_bf(ap, bn3, o3);
    __syncthreads();
    bk0 = nk0; bk1 = nk1;
    bn0 = nn0; bn1 = nn1; bn2 = nn2; bn3 = nn3;
  }
#pragma unroll
  for (int r = 0; r < 8; ++r) {
    int i = i0 + r + 8 * g;
    if (i < NN) {
      size_t base = (size_t)i * DH;
      node2[base + 0  + n] = o0[r] + nodeIn[base + 0  + n];
      node2[base + 16 + n] = o1[r] + nodeIn[base + 16 + n];
      node2[base + 32 + n] = o2[r] + nodeIn[base + 32 + n];
      node2[base + 48 + n] = o3[r] + nodeIn[base + 48 + n];
    }
  }
  red[lane] = ssq;
  __syncthreads();
  if (lane == 0) {
    float t = 0.0f;
    for (int i = 0; i < 32; ++i) t += red[i];
    atomicAdd(sumsq, t);
  }
}

// thr = mean + 0.2*std(ddof=1); mean = 1/N exactly (columns sum to 1)
__global__ void finalize_thr(const float* __restrict__ sumsq,
                             float* __restrict__ thr) {
  double ssq = (double)*sumsq;
  double M = (double)NN * (double)NN;
  double var = (ssq - 1.0) / (M - 1.0);
  if (var < 0.0) var = 0.0;
  *thr = (float)(1.0 / (double)NN + (double)STDF * sqrt(var));
}

// ---- layernorm -------------------------------------------------------------
__global__ __launch_bounds__(256) void layernorm_k(
    const float* __restrict__ X, const float* __restrict__ gamma,
    const float* __restrict__ beta, float* __restrict__ Y) {
  __shared__ float red[256];
  const int row = blockIdx.x, t = threadIdx.x;
  float x = X[(size_t)row * DIMF + t];
  red[t] = x;
  __syncthreads();
  for (int s = 128; s > 0; s >>= 1) { if (t < s) red[t] += red[t + s]; __syncthreads(); }
  float mu = red[0] / DIMF;
  __syncthreads();
  float dx = x - mu;
  red[t] = dx * dx;
  __syncthreads();
  for (int s = 128; s > 0; s >>= 1) { if (t < s) red[t] += red[t + s]; __syncthreads(); }
  float var = red[0] / DIMF;
  Y[(size_t)row * DIMF + t] = dx * rsqrtf(var + LN_EPS) * gamma[t] + beta[t];
}

// ---- adj: recompute attn2, threshold, stream int32 adjacency ---------------
__global__ __launch_bounds__(32) void adj_kernel(
    const float* __restrict__ q2, const bf16_t* __restrict__ k2f,
    const float* __restrict__ colmax, const float* __restrict__ colsum,
    const float* __restrict__ thrPtr, int* __restrict__ adj) {
  const int lane = threadIdx.x & 31;
  const int n = lane & 15, g = lane >> 4;
  const int i0 = blockIdx.x * 16;
  constexpr int JT = NNP / 16;
  const float thr = *thrPtr;
  v16bf aq = load_a_rm(q2, D2, i0, 0);
  v16bf b = load_frag(k2f, 0);
  for (int jt = 0; jt < JT; ++jt) {
    const int jn = (jt + 1 < JT) ? jt + 1 : jt;
    v16bf nb = load_frag(k2f, jn);
    int jp = (jt + 8 < JT) ? jt + 8 : jt;
    __builtin_prefetch(k2f + (size_t)jp * 32 * 16, 0, 1);  // global_prefetch_b8
    v8f d = {};
    d = wmma_bf(aq, b, d);
    const int j = jt * 16 + n;
    const float cm = colmax[j], cs = colsum[j];
    if (j < NN) {
#pragma unroll
      for (int r = 0; r < 8; ++r) {
        int i = i0 + r + 8 * g;
        if (i < NN) {
          float p = __expf(d[r] * SCALE - cm) / cs;
          adj[(size_t)i * NN + j] = (p > thr) ? 1 : 0;
        }
      }
    }
    b = nb;
  }
}

// ---- init: zero score/sumsq and all padded matrices ------------------------
__global__ void init_ws_k(unsigned* __restrict__ scoreBits, float* __restrict__ sumsq,
                          float* __restrict__ node, float* __restrict__ q2,
                          float* __restrict__ k2, float* __restrict__ node2) {
  size_t i = (size_t)blockIdx.x * blockDim.x + threadIdx.x;
  size_t stride = (size_t)gridDim.x * blockDim.x;
  for (size_t t = i; t < NKN; t += stride) scoreBits[t] = 0u;
  for (size_t t = i; t < (size_t)NNP * DH; t += stride) { node[t] = 0.0f; node2[t] = 0.0f; }
  for (size_t t = i; t < (size_t)NNP * D2; t += stride) { q2[t] = 0.0f; k2[t] = 0.0f; }
  if (i == 0) *sumsq = 0.0f;
}

// ---------------------------------------------------------------------------
extern "C" void kernel_launch(void* const* d_in, const int* in_sizes, int n_in,
                              void* d_out, int out_size, void* d_ws, size_t ws_size,
                              hipStream_t stream) {
  (void)in_sizes; (void)n_in; (void)out_size; (void)ws_size;
  const float* q    = (const float*)d_in[0];
  const float* k    = (const float*)d_in[1];
  const float* qW1  = (const float*)d_in[2];  const float* qb1  = (const float*)d_in[3];
  const float* qW2  = (const float*)d_in[4];  const float* qb2  = (const float*)d_in[5];
  const float* kW1  = (const float*)d_in[6];  const float* kb1  = (const float*)d_in[7];
  const float* kW2  = (const float*)d_in[8];  const float* kb2  = (const float*)d_in[9];
  const float* q2W1 = (const float*)d_in[10]; const float* q2b1 = (const float*)d_in[11];
  const float* q2W2 = (const float*)d_in[12]; const float* q2b2 = (const float*)d_in[13];
  const float* k2W1 = (const float*)d_in[14]; const float* k2b1 = (const float*)d_in[15];
  const float* k2W2 = (const float*)d_in[16]; const float* k2b2 = (const float*)d_in[17];
  const float* oW1  = (const float*)d_in[18]; const float* ob1  = (const float*)d_in[19];
  const float* oW2  = (const float*)d_in[20]; const float* ob2  = (const float*)d_in[21];
  const float* gamma= (const float*)d_in[22]; const float* beta = (const float*)d_in[23];

  float* W = (float*)d_ws;
  size_t off = 0;
  auto alloc = [&](size_t nf) { float* p = W + off; off += (nf + 63) & ~(size_t)63; return p; };
  auto allocb = [&](size_t nbf) {
    bf16_t* p = (bf16_t*)(W + off); off += ((nbf + 1) / 2 + 63) & ~(size_t)63; return p;
  };
  float*  qh     = alloc((size_t)NQ * DH);
  float*  kh     = alloc((size_t)NKN * DH);
  float*  rowmax = alloc(NQ);
  float*  rowsum = alloc(NQ);
  float*  score  = alloc(NKN);
  float*  node   = alloc((size_t)NNP * DH);
  float*  q2     = alloc((size_t)NNP * D2);
  float*  k2     = alloc((size_t)NNP * D2);
  float*  colmax = alloc(NNP);
  float*  colsum = alloc(NNP);
  float*  node2  = alloc((size_t)NNP * DH);
  float*  node3  = alloc((size_t)NN * DIMF);
  float*  sumsq  = alloc(1);
  float*  thr    = alloc(1);
  bf16_t* khf    = allocb((size_t)NKN * DH);        // kh^T fragments
  bf16_t* k2f    = allocb((size_t)NNP * D2);        // k2^T fragments
  bf16_t* nodef  = allocb((size_t)NNP * DH);        // node  fragments (KN)
  bf16_t* qW1f   = allocb(256 * 128);
  bf16_t* qW2f   = allocb(128 * 64);
  bf16_t* kW1f   = allocb(256 * 128);
  bf16_t* kW2f   = allocb(128 * 64);
  bf16_t* q2W1f  = allocb(64 * 32);
  bf16_t* q2W2f  = allocb(32 * 32);
  bf16_t* k2W1f  = allocb(64 * 32);
  bf16_t* k2W2f  = allocb(32 * 32);
  bf16_t* oW1f   = allocb(64 * 128);
  bf16_t* oW2f   = allocb(128 * 256);

  float* outNode = (float*)d_out;
  int*   outAdj  = (int*)((float*)d_out + (size_t)NN * DIMF);

  const int qT = NQ / 16;          // 500
  const int nT = NNP / 16;         // 702

  init_ws_k<<<512, 256, 0, stream>>>((unsigned*)score, sumsq, node, q2, k2, node2);

  pack_b_kn<<<8 * 8,  32, 0, stream>>>(qW1, 256, 128, qW1f);
  pack_b_kn<<<4 * 4,  32, 0, stream>>>(qW2, 128, 64,  qW2f);
  pack_b_kn<<<8 * 8,  32, 0, stream>>>(kW1, 256, 128, kW1f);
  pack_b_kn<<<4 * 4,  32, 0, stream>>>(kW2, 128, 64,  kW2f);
  pack_b_kn<<<2 * 2,  32, 0, stream>>>(q2W1, 64, 32,  q2W1f);
  pack_b_kn<<<1 * 2,  32, 0, stream>>>(q2W2, 32, 32,  q2W2f);
  pack_b_kn<<<2 * 2,  32, 0, stream>>>(k2W1, 64, 32,  k2W1f);
  pack_b_kn<<<1 * 2,  32, 0, stream>>>(k2W2, 32, 32,  k2W2f);
  pack_b_kn<<<2 * 8,  32, 0, stream>>>(oW1, 64, 128,  oW1f);
  pack_b_kn<<<4 * 16, 32, 0, stream>>>(oW2, 128, 256, oW2f);

  mlp2_wmma<256, 128, 64><<<qT, 32, 0, stream>>>(q, NQ,  qW1f, qb1, qW2f, qb2, qh);
  mlp2_wmma<256, 128, 64><<<qT, 32, 0, stream>>>(k, NKN, kW1f, kb1, kW2f, kb2, kh);

  pack_b_nk<<<(NKN / 16) * 2, 32, 0, stream>>>(kh, DH, khf);
  attn1_rowstats<<<qT, 32, 0, stream>>>(qh, khf, rowmax, rowsum);
  attn1_score<<<qT, 32, 0, stream>>>(qh, khf, rowmax, rowsum, (unsigned*)score);

  copy_qh<<<(NQ * DH + 255) / 256, 256, 0, stream>>>(qh, node);
  select_topk<<<(NKN + 255) / 256, 256, 0, stream>>>(score, kh, node);

  mlp2_wmma<64, 32, 32><<<nT, 32, 0, stream>>>(node, NN, q2W1f, q2b1, q2W2f, q2b2, q2);
  mlp2_wmma<64, 32, 32><<<nT, 32, 0, stream>>>(node, NN, k2W1f, k2b1, k2W2f, k2b2, k2);

  pack_b_nk<<<NNP / 16, 32, 0, stream>>>(k2, D2, k2f);            // KT=1
  pack_b_kn<<<(NNP / 32) * 4, 32, 0, stream>>>(node, NNP, DH, nodef);

  attn2_colstats<<<nT, 32, 0, stream>>>(q2, k2f, colmax, colsum);
  attn2_apply<<<nT, 32, 0, stream>>>(q2, k2f, nodef, node, colmax, colsum, node2, sumsq);
  finalize_thr<<<1, 1, 0, stream>>>(sumsq, thr);

  mlp2_wmma<64, 128, 256><<<nT, 32, 0, stream>>>(node2, NN, oW1f, ob1, oW2f, ob2, node3);
  layernorm_k<<<NN, 256, 0, stream>>>(node3, gamma, beta, outNode);

  adj_kernel<<<nT, 32, 0, stream>>>(q2, k2f, colmax, colsum, thr, outAdj);
}
